// MultiheadAttention_22230750724131
// MI455X (gfx1250) — matrix operand order repaired and checked by reading, not compiled, not run on hardware
//
#include <hip/hip_runtime.h>
#include <hip/hip_bf16.h>

// ---------------------------------------------------------------------------
// MHA forward for MI455X (gfx1250, wave32, WMMA).
//   B=2, S=2048, DIN=1024, D=1024, H=16, DK=64.
// Pipeline (f16 compute, f32 accumulate via v_wmma_f32_16x16x32_f16):
//   1) cvt f32->f16 for x, w_qkv, w_o
//   2) QKV GEMM (NT, 32x64 wave tiles): qkv = x @ w_qkv^T + b_qkv, scattered as
//        Q (b,h,s,dk) PRE-SCALED by 1/sqrt(dk), K (b,h,s,dk), V^T (b,h,dk,s)
//   3) flash attention: per-wave 16-query tile, stream keys in chunks of 32,
//        online softmax, P relayout via per-wave LDS (C-layout -> A-layout)
//   4) out GEMM (NT, 32x64 wave tiles): y = attn @ w_o^T + b_o  (f32 out)
// ~50MB working set fits the 192MB L2; GEMM operand re-reads are L2 hits, so
// the kernels are sized to keep the WMMA pipe busy (8 WMMAs / 12 b128 loads).
// ---------------------------------------------------------------------------

typedef _Float16 half_t;
typedef __attribute__((ext_vector_type(16))) _Float16 v16h;
typedef __attribute__((ext_vector_type(8)))  _Float16 v8h;
typedef __attribute__((ext_vector_type(4)))  _Float16 v4h;
typedef __attribute__((ext_vector_type(8)))  float    v8f;
typedef __attribute__((ext_vector_type(4)))  float    v4f;

#define NB   2
#define NS   2048
#define NDIN 1024
#define ND   1024
#define NH   16
#define NDK  64
#define N3D  3072
#define NM   (NB * NS)          // 4096 token rows
static constexpr float ATT_SCALE = 0.125f;   // 1/sqrt(64)

// ---------------- fragment loaders (contiguous k = 16h + s in both A and B) -------

// A fragment, 16x32 f16 (MxK), source row-major with given stride (halves).
// lanes 0-15: rows 0-15, K = {0..15}; lanes 16-31: rows 0-15, K = {16..31}.
__device__ __forceinline__ v16h load_a_frag(const half_t* __restrict__ src,
                                            int stride, int m0, int k0, int lane) {
  const int row = m0 + (lane & 15);
  const int g   = lane >> 4;
  const half_t* p = src + (size_t)row * stride + k0 + 16 * g;
  v8h lo = *(const v8h*)p;
  v8h hi = *(const v8h*)(p + 8);
  v16h r;
#pragma unroll
  for (int i = 0; i < 8; ++i) { r[i] = lo[i]; r[i + 8] = hi[i]; }
  return r;
}

// B fragment, 32x16 f16 (KxN) of B = W^T, where W is row-major (N x K, stride halves).
// lane n (col n&15), group g: holds B[16g..16g+15, n] = W[n0+n, k0+16g .. +15] (contiguous).
__device__ __forceinline__ v16h load_b_frag_nt(const half_t* __restrict__ w,
                                               int stride, int n0, int k0, int lane) {
  const int n = n0 + (lane & 15);
  const int g = lane >> 4;
  return *(const v16h*)(w + (size_t)n * stride + k0 + 16 * g);
}

__device__ __forceinline__ v8f wmma_f16f32(v16h a, v16h b, v8f c) {
  return __builtin_amdgcn_wmma_f32_16x16x32_f16(false, a, false, b, (short)0, c,
                                                false, false);
}

// Shared 32x64 wave-tile NT GEMM inner loop: 8 WMMAs per 32-k step,
// A fragments reused 4x, B fragments reused 2x.
template <int KDIM>
__device__ __forceinline__ void gemm_core_32x64(const half_t* __restrict__ A,
                                                const half_t* __restrict__ W,
                                                int m0, int n0, int lane,
                                                v8f acc[2][4]) {
  for (int k0 = 0; k0 < KDIM; k0 += 32) {
    v16h a0 = load_a_frag(A, KDIM, m0,      k0, lane);
    v16h a1 = load_a_frag(A, KDIM, m0 + 16, k0, lane);
#pragma unroll
    for (int f = 0; f < 4; ++f) {
      v16h b = load_b_frag_nt(W, KDIM, n0 + 16 * f, k0, lane);
      acc[0][f] = wmma_f16f32(a0, b, acc[0][f]);
      acc[1][f] = wmma_f16f32(a1, b, acc[1][f]);
    }
  }
}

// ---------------- kernel 1: f32 -> f16 convert ------------------------------

__global__ void cvt_f32_f16(const float* __restrict__ src, half_t* __restrict__ dst,
                            int n4) {
  int i = blockIdx.x * blockDim.x + threadIdx.x;
  if (i < n4) {
    v4f v = *(const v4f*)(src + 4 * (size_t)i);
    v4h h;
#pragma unroll
    for (int j = 0; j < 4; ++j) h[j] = (half_t)v[j];
    *(v4h*)(dst + 4 * (size_t)i) = h;
  }
}

// ---------------- kernel 2: QKV GEMM with scatter ---------------------------
// grid (NM/64, N3D/256), block 256 (8 waves, 2 along m x 4 along n).
// Q is stored pre-scaled by 1/sqrt(dk) so flash attention skips the multiply.

__global__ void __launch_bounds__(256, 2)
gemm_qkv(const half_t* __restrict__ X, const half_t* __restrict__ W,
         const float* __restrict__ bias, half_t* __restrict__ Qb,
         half_t* __restrict__ Kb, half_t* __restrict__ VTb) {
  const int lane = threadIdx.x & 31;
  const int wave = threadIdx.x >> 5;
  const int m0 = blockIdx.x * 64 + (wave & 1) * 32;
  const int n0 = blockIdx.y * 256 + (wave >> 1) * 64;

  v8f acc[2][4] = {};
  gemm_core_32x64<NDIN>(X, W, m0, n0, lane, acc);

  const int g = lane >> 4, nl = lane & 15;
#pragma unroll
  for (int mt = 0; mt < 2; ++mt) {
#pragma unroll
    for (int f = 0; f < 4; ++f) {
      const int n = n0 + 16 * f + nl;
      const float bn = bias[n];
      const int sec = n >> 10;           // 0=Q 1=K 2=V
      const int c = n & 1023;
      const int h = c >> 6, d = c & 63;
#pragma unroll
      for (int r = 0; r < 8; ++r) {
        const int m = m0 + 16 * mt + r + 8 * g;
        const int b = m >> 11, s = m & 2047;
        float v = acc[mt][f][r] + bn;
        const size_t bh = (size_t)b * NH + h;
        if (sec == 0)      Qb [(bh * NS + s) * NDK + d] = (half_t)(v * ATT_SCALE);
        else if (sec == 1) Kb [(bh * NS + s) * NDK + d] = (half_t)v;
        else               VTb[(bh * NDK + d) * NS + s] = (half_t)v;  // V^T
      }
    }
  }
}

// ---------------- kernel 3: flash attention ---------------------------------
// grid (NS/128, NB*NH), block 256. Each wave: one 16-query tile, streams all
// 2048 keys in chunks of 32. LDS stride padded to 40 halves (bank spread).

#define PLDS_STRIDE 40

__global__ void __launch_bounds__(256, 2)
flash_attn(const half_t* __restrict__ Qb, const half_t* __restrict__ Kb,
           const half_t* __restrict__ VTb, half_t* __restrict__ Ob) {
  __shared__ __align__(16) half_t plds[8 * 16 * PLDS_STRIDE];
  const int lane = threadIdx.x & 31;
  const int wave = threadIdx.x >> 5;
  const int bh = blockIdx.y;
  const int b = bh >> 4, h = bh & 15;
  const int m0 = blockIdx.x * 128 + wave * 16;

  const half_t* Q  = Qb  + (size_t)bh * NS * NDK;
  const half_t* K  = Kb  + (size_t)bh * NS * NDK;
  const half_t* VT = VTb + (size_t)bh * NDK * NS;
  half_t* lw = plds + wave * 16 * PLDS_STRIDE;

  const int g = lane >> 4, nl = lane & 15;

  // Q tile resident in registers: 16x64 = two A fragments (already scaled).
  const v16h aq0 = load_a_frag(Q, NDK, m0, 0,  lane);
  const v16h aq1 = load_a_frag(Q, NDK, m0, 32, lane);

  v8f o[4] = {};
  v8f mi, li;
#pragma unroll
  for (int r = 0; r < 8; ++r) { mi[r] = -3.0e38f; li[r] = 0.f; }

  for (int kc = 0; kc < NS; kc += 32) {
    // logits: 16 queries x 32 keys (two 16x16 C fragments), dk contraction.
    v8f s0 = {}, s1 = {};
    s0 = wmma_f16f32(aq0, load_b_frag_nt(K, NDK, kc,      0,  lane), s0);
    s0 = wmma_f16f32(aq1, load_b_frag_nt(K, NDK, kc,      32, lane), s0);
    s1 = wmma_f16f32(aq0, load_b_frag_nt(K, NDK, kc + 16, 0,  lane), s1);
    s1 = wmma_f16f32(aq1, load_b_frag_nt(K, NDK, kc + 16, 32, lane), s1);

    v8f t;
#pragma unroll
    for (int r = 0; r < 8; ++r) t[r] = fmaxf(s0[r], s1[r]);
    // row max across 16 lanes of this half-wave (xor masks stay in-half).
#pragma unroll
    for (int off = 1; off < 16; off <<= 1)
#pragma unroll
      for (int r = 0; r < 8; ++r) t[r] = fmaxf(t[r], __shfl_xor(t[r], off, 32));

    v8f alpha, p0, p1, rs;
#pragma unroll
    for (int r = 0; r < 8; ++r) {
      const float mnew = fmaxf(mi[r], t[r]);
      alpha[r] = __expf(mi[r] - mnew);
      p0[r] = __expf(s0[r] - mnew);
      p1[r] = __expf(s1[r] - mnew);
      rs[r] = p0[r] + p1[r];
      mi[r] = mnew;
    }
#pragma unroll
    for (int off = 1; off < 16; off <<= 1)
#pragma unroll
      for (int r = 0; r < 8; ++r) rs[r] += __shfl_xor(rs[r], off, 32);
#pragma unroll
    for (int r = 0; r < 8; ++r) li[r] = li[r] * alpha[r] + rs[r];
#pragma unroll
    for (int f = 0; f < 4; ++f)
#pragma unroll
      for (int r = 0; r < 8; ++r) o[f][r] *= alpha[r];

    // P: C-layout -> LDS (16x32 row-major, padded) -> A-layout.
#pragma unroll
    for (int r = 0; r < 8; ++r) {
      const int row = r + 8 * g;
      lw[row * PLDS_STRIDE + nl]      = (half_t)p0[r];
      lw[row * PLDS_STRIDE + 16 + nl] = (half_t)p1[r];
    }
    asm volatile("s_wait_dscnt 0" ::: "memory");   // same-wave DS RAW
    const half_t* pp = lw + nl * PLDS_STRIDE + 16 * g;
    v8h plo = *(const v8h*)pp;
    v8h phi = *(const v8h*)(pp + 8);
    v16h pa;
#pragma unroll
    for (int i = 0; i < 8; ++i) { pa[i] = plo[i]; pa[i + 8] = phi[i]; }

    // O += P @ V : contraction over 32 keys, V^T gives contiguous B columns.
#pragma unroll
    for (int f = 0; f < 4; ++f)
      o[f] = wmma_f16f32(pa, load_b_frag_nt(VT, NS, 16 * f, kc, lane), o[f]);
  }

  // epilogue: normalize and write attn output (b, s, h*64 + d) as f16.
  half_t* outp = Ob + (size_t)b * NS * ND + (size_t)h * NDK;
#pragma unroll
  for (int f = 0; f < 4; ++f) {
    const int d = 16 * f + nl;
#pragma unroll
    for (int r = 0; r < 8; ++r) {
      const int q = m0 + r + 8 * g;
      outp[(size_t)q * ND + d] = (half_t)(o[f][r] / li[r]);
    }
  }
}

// ---------------- kernel 4: output projection (f32 result) -----------------
// grid (NM/64, ND/256), block 256 (8 waves, 2 along m x 4 along n).

__global__ void __launch_bounds__(256, 2)
gemm_out(const half_t* __restrict__ A, const half_t* __restrict__ W,
         const float* __restrict__ bias, float* __restrict__ Y) {
  const int lane = threadIdx.x & 31;
  const int wave = threadIdx.x >> 5;
  const int m0 = blockIdx.x * 64 + (wave & 1) * 32;
  const int n0 = blockIdx.y * 256 + (wave >> 1) * 64;

  v8f acc[2][4] = {};
  gemm_core_32x64<ND>(A, W, m0, n0, lane, acc);

  const int g = lane >> 4, nl = lane & 15;
#pragma unroll
  for (int mt = 0; mt < 2; ++mt) {
#pragma unroll
    for (int f = 0; f < 4; ++f) {
      const int n = n0 + 16 * f + nl;
      const float bn = bias[n];
#pragma unroll
      for (int r = 0; r < 8; ++r) {
        const int m = m0 + 16 * mt + r + 8 * g;
        Y[(size_t)m * ND + n] = acc[mt][f][r] + bn;
      }
    }
  }
}

// ---------------- host launcher ---------------------------------------------

extern "C" void kernel_launch(void* const* d_in, const int* in_sizes, int n_in,
                              void* d_out, int out_size, void* d_ws, size_t ws_size,
                              hipStream_t stream) {
  const float* x     = (const float*)d_in[0];
  const float* w_qkv = (const float*)d_in[1];
  const float* b_qkv = (const float*)d_in[2];
  const float* w_o   = (const float*)d_in[3];
  const float* b_o   = (const float*)d_in[4];
  float* y = (float*)d_out;

  // workspace carve-up (f16 buffers), ~48 MiB total
  char* ws = (char*)d_ws;
  size_t off = 0;
  auto take = [&](size_t bytes) { char* p = ws + off; off += (bytes + 255) & ~(size_t)255; return p; };
  half_t* x_h    = (half_t*)take((size_t)NM * NDIN * 2);
  half_t* wqkv_h = (half_t*)take((size_t)N3D * NDIN * 2);
  half_t* wo_h   = (half_t*)take((size_t)ND * ND * 2);
  half_t* Qb     = (half_t*)take((size_t)NB * NH * NS * NDK * 2);
  half_t* Kb     = (half_t*)take((size_t)NB * NH * NS * NDK * 2);
  half_t* VTb    = (half_t*)take((size_t)NB * NH * NDK * NS * 2);
  half_t* attn_h = (half_t*)take((size_t)NM * ND * 2);
  (void)ws_size;

  // 1) converts
  {
    int n4;
    n4 = (NM * NDIN) / 4;
    cvt_f32_f16<<<(n4 + 255) / 256, 256, 0, stream>>>(x, x_h, n4);
    n4 = (N3D * NDIN) / 4;
    cvt_f32_f16<<<(n4 + 255) / 256, 256, 0, stream>>>(w_qkv, wqkv_h, n4);
    n4 = (ND * ND) / 4;
    cvt_f32_f16<<<(n4 + 255) / 256, 256, 0, stream>>>(w_o, wo_h, n4);
  }
  // 2) QKV projection + scatter (Q pre-scaled)
  gemm_qkv<<<dim3(NM / 64, N3D / 256), 256, 0, stream>>>(x_h, wqkv_h, b_qkv,
                                                         Qb, Kb, VTb);
  // 3) flash attention
  flash_attn<<<dim3(NS / 128, NB * NH), 256, 0, stream>>>(Qb, Kb, VTb, attn_h);
  // 4) output projection
  gemm_out<<<dim3(NM / 64, ND / 256), 256, 0, stream>>>(attn_h, wo_h, b_o, y);
}
